// MultiScaleBlockTPV_19731079757887
// MI455X (gfx1250) — matrix operand order repaired and checked by reading, not compiled, not run on hardware
//
#include <hip/hip_runtime.h>
#include <hip/hip_fp16.h>
#include <math.h>

// ---------------------------------------------------------------------------
// MultiScaleBlockTPV for MI455X (gfx1250):
//   wave32 + v_wmma_f32_16x16x32_f16 + global_load_async_to_lds_b128 staging
// ---------------------------------------------------------------------------

typedef __attribute__((ext_vector_type(16))) _Float16 v16h;
typedef __attribute__((ext_vector_type(8)))  float    v8f;

#define N_TOK 4096
#define CDIM  256
#define NHEAD 8
#define HD    32
#define FFND  512
#define QKVD  768

// issue one async 16-byte global->LDS DMA (ASYNCcnt-tracked)
__device__ __forceinline__ void async_b128(unsigned lds_off, unsigned long long gaddr) {
  asm volatile("global_load_async_to_lds_b128 %0, %1, off"
               :: "v"(lds_off), "v"(gaddr) : "memory");
}
__device__ __forceinline__ void wait_async0() {
  asm volatile("s_wait_asynccnt 0x0" ::: "memory");
}

// ---------------------------------------------------------------------------
// elementwise f32 -> f16 convert
// ---------------------------------------------------------------------------
__global__ void k_f16(const float* __restrict__ s, _Float16* __restrict__ d, int n) {
  int i = blockIdx.x * blockDim.x + threadIdx.x;
  if (i < n) d[i] = (_Float16)s[i];
}

// x [C, 4096] -> xT [4096, C]
__global__ void k_transpose(const float* __restrict__ x, float* __restrict__ xT) {
  int n = blockIdx.x, c = threadIdx.x;
  xT[(size_t)n * CDIM + c] = x[(size_t)c * N_TOK + n];
}

// per-split carve permutation: slot (blk, j) -> spatial token (z*16+h)*16+w
__global__ void k_build_idx(int* __restrict__ idxall) {
  int gi = blockIdx.x * blockDim.x + threadIdx.x;      // 0 .. 4*4096-1
  int split = gi >> 12, t = gi & 4095;
  int s = 1 << split;                                  // 1,2,4,8
  int bsz = 16 / s, bs = bsz * bsz * bsz;
  int blk = t / bs, j = t % bs;
  int xb = blk % s, yb = (blk / s) % s, zb = blk / (s * s);
  int xi = j % bsz, yi = (j / bsz) % bsz, zi = j / (bsz * bsz);
  int z = zb * bsz + zi, h = yb * bsz + yi, w = xb * bsz + xi;
  idxall[gi] = (z * 16 + h) * 16 + w;
}

// ---------------------------------------------------------------------------
// WMMA GEMM:  C[M,N] = act(A[M,K] @ B[N,K]^T + bias) + res
//   4 waves / block; per K-step the 64x32h B panel is DMA'd to LDS once
//   (async-to-LDS) and shared by all 4 waves; each wave owns 16 rows x 64 cols.
//   act: 0 = none, 1 = exact GELU
// ---------------------------------------------------------------------------
__global__ __launch_bounds__(128) void k_gemm(
    const _Float16* __restrict__ A, const _Float16* __restrict__ B,
    const float* __restrict__ bias, const float* __restrict__ res,
    float* __restrict__ Cout, int M, int N, int K, int act)
{
  __shared__ _Float16 Blds[64 * 32];
  const int tid  = threadIdx.x;
  const int lane = tid & 31;
  const int wave = tid >> 5;
  const int row0 = blockIdx.x * 64 + wave * 16;
  const int n0   = blockIdx.y * 64;
  const int m = lane & 15, hi = lane >> 4;
  const unsigned ldsbase = (unsigned)(size_t)&Blds[0];

  v8f acc[4] = {};

  int arow = row0 + m; if (arow >= M) arow = M - 1;
  const _Float16* Arow = A + (size_t)arow * K;

  for (int k0 = 0; k0 < K; k0 += 32) {
    // ---- async DMA: B tile [64 rows x 32 halfs] -> LDS (256 x 16B chunks) ----
    #pragma unroll
    for (int e = 0; e < 2; ++e) {
      int ci = tid + e * 128;                 // 0..255
      int brow = ci >> 2, bchk = ci & 3;      // row 0..63, 16B chunk 0..3
      unsigned long long ga = (unsigned long long)(size_t)
          (B + (size_t)(n0 + brow) * K + k0 + bchk * 8);
      async_b128(ldsbase + (unsigned)((brow * 32 + bchk * 8) * 2), ga);
    }
    __builtin_prefetch((const void*)(Arow + k0 + 128), 0, 1);
    // ---- A fragment direct from global (per-wave rows) ----
    v16h a;
    {
      const _Float16* pa = Arow + k0 + 8 * hi;
      #pragma unroll
      for (int i = 0; i < 8; ++i) { a[i] = pa[i]; a[8 + i] = pa[16 + i]; }
    }
    wait_async0();
    __syncthreads();
    // ---- 4 WMMAs per wave against the shared LDS B panel ----
    #pragma unroll
    for (int t = 0; t < 4; ++t) {
      const _Float16* pb = &Blds[(t * 16 + m) * 32 + 8 * hi];
      v16h b;
      #pragma unroll
      for (int i = 0; i < 8; ++i) { b[i] = pb[i]; b[8 + i] = pb[16 + i]; }
      acc[t] = __builtin_amdgcn_wmma_f32_16x16x32_f16(false, a, false, b,
                                                      (short)0, acc[t], false, false);
    }
    __syncthreads();
  }

  #pragma unroll
  for (int t = 0; t < 4; ++t) {
    int col = n0 + t * 16 + m;
    float bv = bias ? bias[col] : 0.f;
    #pragma unroll
    for (int r = 0; r < 8; ++r) {
      int row = row0 + r + 8 * hi;
      if (row < M) {
        float v = acc[t][r] + bv;
        if (act == 1) v = 0.5f * v * (1.f + erff(v * 0.70710678118654752f));
        if (res) v += res[(size_t)row * N + col];
        Cout[(size_t)row * N + col] = v;
      }
    }
  }
}

// ---------------------------------------------------------------------------
// Flash-style wave32 attention over one (block, head, 16-query tile).
// qkv: [T, 768] f16 with q|k|v at col offsets 0|256|512. idx==nullptr -> identity.
// out: [T, 256] f32 (head-major cols).
// ---------------------------------------------------------------------------
__global__ __launch_bounds__(32) void k_attn(
    const _Float16* __restrict__ qkv, const int* __restrict__ idx,
    float* __restrict__ out, int bs, float scale)
{
  const int qt = blockIdx.x, head = blockIdx.y, blk = blockIdx.z;
  const int lane = threadIdx.x;
  const int m = lane & 15, hi = lane >> 4;
  __shared__ _Float16 Plds[16 * 32];
  __shared__ _Float16 Vlds[32 * 32];
  const unsigned vldsbase = (unsigned)(size_t)&Vlds[0];

  // Q fragment (A layout: rows = queries, K = head dim = 32)
  int qrow = qt * 16 + m;
  int qslot = blk * bs + (qrow < bs ? qrow : 0);
  int qtok = idx ? idx[qslot] : qslot;
  const _Float16* Qp = qkv + (size_t)qtok * QKVD + head * HD + 8 * hi;
  v16h qa;
  #pragma unroll
  for (int i = 0; i < 8; ++i) { qa[i] = Qp[i]; qa[8 + i] = Qp[16 + i]; }

  float mrow[8], lrow[8];
  v8f o0 = {}, o1 = {};
  #pragma unroll
  for (int r = 0; r < 8; ++r) { mrow[r] = -1e30f; lrow[r] = 0.f; }

  const int nkc = (bs + 31) / 32;
  for (int kc = 0; kc < nkc; ++kc) {
    // ---- async DMA: V chunk [32 keys x 32 halfs] -> LDS (4 x 16B per lane) ----
    // masked keys read a clamped (valid) token; their P columns are 0 anyway.
    {
      int key = kc * 32 + lane;
      int kslot = blk * bs + (key < bs ? key : 0);
      int vtok = idx ? idx[kslot] : kslot;
      unsigned long long ga = (unsigned long long)(size_t)
          (qkv + (size_t)vtok * QKVD + 2 * CDIM + head * HD);
      unsigned la = vldsbase + (unsigned)(lane * 32 * 2);
      #pragma unroll
      for (int ch = 0; ch < 4; ++ch)
        async_b128(la + (unsigned)(ch * 16), ga + (unsigned long long)(ch * 16));
    }
    // ---- scores: two 16x16 tiles, contraction over D=32 ----
    v8f sc[2];
    #pragma unroll
    for (int s = 0; s < 2; ++s) {
      int key = kc * 32 + s * 16 + m;
      int kslot = blk * bs + (key < bs ? key : 0);
      int ktok = idx ? idx[kslot] : kslot;
      const _Float16* Kp = qkv + (size_t)ktok * QKVD + CDIM + head * HD + 8 * hi;
      v16h kb;
      #pragma unroll
      for (int i = 0; i < 8; ++i) { kb[i] = Kp[i]; kb[8 + i] = Kp[16 + i]; }
      v8f z = {};
      sc[s] = __builtin_amdgcn_wmma_f32_16x16x32_f16(false, qa, false, kb,
                                                     (short)0, z, false, false);
    }
    // ---- online softmax (rows live in 16-lane groups) ----
    float p0[8], p1[8];
    int key0 = kc * 32 + m, key1 = kc * 32 + 16 + m;
    #pragma unroll
    for (int r = 0; r < 8; ++r) {
      float v0 = (key0 < bs) ? sc[0][r] * scale : -1e30f;
      float v1 = (key1 < bs) ? sc[1][r] * scale : -1e30f;
      float cm = fmaxf(v0, v1);
      #pragma unroll
      for (int off = 1; off < 16; off <<= 1) cm = fmaxf(cm, __shfl_xor(cm, off, 32));
      float mn = fmaxf(mrow[r], cm);
      float al = __expf(mrow[r] - mn);
      float e0 = __expf(v0 - mn);
      float e1 = __expf(v1 - mn);
      float rs = e0 + e1;
      #pragma unroll
      for (int off = 1; off < 16; off <<= 1) rs += __shfl_xor(rs, off, 32);
      lrow[r] = lrow[r] * al + rs;
      mrow[r] = mn;
      o0[r] *= al; o1[r] *= al;
      p0[r] = e0; p1[r] = e1;
    }
    // ---- stage P (C-layout -> LDS) ----
    #pragma unroll
    for (int r = 0; r < 8; ++r) {
      int row = r + 8 * hi;
      Plds[row * 32 + m]      = (_Float16)p0[r];
      Plds[row * 32 + 16 + m] = (_Float16)p1[r];
    }
    wait_async0();
    __syncthreads();
    // P re-read in A-fragment layout
    v16h pa;
    {
      const _Float16* Pp = &Plds[m * 32 + 8 * hi];
      #pragma unroll
      for (int i = 0; i < 8; ++i) { pa[i] = Pp[i]; pa[8 + i] = Pp[16 + i]; }
    }
    // V as two B fragments (d 0-15, d 16-31), contraction = 32 keys
    v16h vb0, vb1;
    #pragma unroll
    for (int i = 0; i < 16; ++i) {
      int krow = 8 * hi + (i & 7) + ((i >> 3) << 4);
      vb0[i] = Vlds[krow * 32 + m];
      vb1[i] = Vlds[krow * 32 + 16 + m];
    }
    o0 = __builtin_amdgcn_wmma_f32_16x16x32_f16(false, pa, false, vb0, (short)0, o0, false, false);
    o1 = __builtin_amdgcn_wmma_f32_16x16x32_f16(false, pa, false, vb1, (short)0, o1, false, false);
    __syncthreads();
  }
  // ---- epilogue: O /= l, scatter back through idx ----
  #pragma unroll
  for (int r = 0; r < 8; ++r) {
    int row = qt * 16 + r + 8 * hi;
    if (row < bs) {
      float inv = 1.f / lrow[r];
      int slot = blk * bs + row;
      int tok = idx ? idx[slot] : slot;
      float* op = out + (size_t)tok * CDIM + head * HD;
      op[m]      = o0[r] * inv;
      op[16 + m] = o1[r] * inv;
    }
  }
}

// ---------------------------------------------------------------------------
// out[row] = LN(a[row] + res[row]) * g + b      (one 256-thread block per row)
// ---------------------------------------------------------------------------
__global__ __launch_bounds__(256) void k_add_ln(
    const float* __restrict__ a, const float* __restrict__ resid,
    const float* __restrict__ g, const float* __restrict__ be,
    float* __restrict__ out)
{
  int row = blockIdx.x, c = threadIdx.x;
  __shared__ float red[256];
  float v = a[(size_t)row * CDIM + c] + resid[(size_t)row * CDIM + c];
  red[c] = v; __syncthreads();
  for (int s = 128; s > 0; s >>= 1) { if (c < s) red[c] += red[c + s]; __syncthreads(); }
  float mean = red[0] * (1.f / CDIM);
  __syncthreads();
  float d = v - mean;
  red[c] = d * d; __syncthreads();
  for (int s = 128; s > 0; s >>= 1) { if (c < s) red[c] += red[c + s]; __syncthreads(); }
  float var = red[0] * (1.f / CDIM);
  out[(size_t)row * CDIM + c] = d * rsqrtf(var + 1e-5f) * g[c] + be[c];
}

// block means: r[blk][c] = mean_j t[idx[blk*bs+j]][c]
__global__ __launch_bounds__(256) void k_block_mean(
    const float* __restrict__ t, const int* __restrict__ idx,
    float* __restrict__ r, int bs)
{
  int blk = blockIdx.x, c = threadIdx.x;
  float s = 0.f;
  for (int j = 0; j < bs; ++j) s += t[(size_t)idx[blk * bs + j] * CDIM + c];
  r[(size_t)blk * CDIM + c] = s / (float)bs;
}

// t[idx[slot]][c] += r2[slot/bs][c]
__global__ __launch_bounds__(256) void k_add_bcast(
    float* __restrict__ t, const float* __restrict__ r2,
    const int* __restrict__ idx, int bs)
{
  int slot = blockIdx.x, c = threadIdx.x;
  int blk = slot / bs;
  int tok = idx[slot];
  t[(size_t)tok * CDIM + c] += r2[(size_t)blk * CDIM + c];
}

// fused16[n][off + c] = f16(t[n][c] + xT[n][c])
__global__ __launch_bounds__(256) void k_to_fused(
    const float* __restrict__ t, const float* __restrict__ xT,
    _Float16* __restrict__ fused16, int off)
{
  int n = blockIdx.x, c = threadIdx.x;
  fused16[(size_t)n * (4 * CDIM) + off + c] =
      (_Float16)(t[(size_t)n * CDIM + c] + xT[(size_t)n * CDIM + c]);
}

// per-channel biased batch stats over 4096 tokens
__global__ __launch_bounds__(256) void k_bn_stats(
    const float* __restrict__ y, float* __restrict__ stats)
{
  int c = blockIdx.x, tid = threadIdx.x;
  __shared__ float s1[256], s2[256];
  float a = 0.f, b = 0.f;
  for (int n = tid; n < N_TOK; n += 256) {
    float v = y[(size_t)n * CDIM + c];
    a += v; b += v * v;
  }
  s1[tid] = a; s2[tid] = b; __syncthreads();
  for (int s = 128; s > 0; s >>= 1) {
    if (tid < s) { s1[tid] += s1[tid + s]; s2[tid] += s2[tid + s]; }
    __syncthreads();
  }
  if (tid == 0) {
    float m = s1[0] * (1.f / N_TOK);
    stats[c] = m;
    stats[CDIM + c] = s2[0] * (1.f / N_TOK) - m * m;
  }
}

// BN + ReLU + transpose back to [C, Z, H, W]
__global__ __launch_bounds__(256) void k_bn_out(
    const float* __restrict__ y, const float* __restrict__ stats,
    const float* __restrict__ g, const float* __restrict__ be,
    float* __restrict__ out)
{
  int c = blockIdx.x;
  int n = blockIdx.y * 256 + threadIdx.x;
  float m = stats[c], var = stats[CDIM + c];
  float v = (y[(size_t)n * CDIM + c] - m) * rsqrtf(var + 1e-5f) * g[c] + be[c];
  out[(size_t)c * N_TOK + n] = fmaxf(v, 0.f);
}

// ---------------------------------------------------------------------------
extern "C" void kernel_launch(void* const* d_in, const int* in_sizes, int n_in,
                              void* d_out, int out_size, void* d_ws, size_t ws_size,
                              hipStream_t stream) {
  (void)in_sizes; (void)n_in; (void)out_size; (void)ws_size;
  const float* X      = (const float*)d_in[0];
  const float* iW_qkv = (const float*)d_in[1];
  const float* iB_qkv = (const float*)d_in[2];
  const float* iW_out = (const float*)d_in[3];
  const float* iB_out = (const float*)d_in[4];
  const float* iLN_g  = (const float*)d_in[5];
  const float* iLN_b  = (const float*)d_in[6];
  const float* eW_qkv = (const float*)d_in[7];
  const float* eB_qkv = (const float*)d_in[8];
  const float* eW_out = (const float*)d_in[9];
  const float* eB_out = (const float*)d_in[10];
  const float* eLN_g  = (const float*)d_in[11];
  const float* eLN_b  = (const float*)d_in[12];
  const float* fW1    = (const float*)d_in[13];
  const float* fB1    = (const float*)d_in[14];
  const float* fW2    = (const float*)d_in[15];
  const float* fB2    = (const float*)d_in[16];
  const float* fuseW  = (const float*)d_in[17];
  const float* bnG    = (const float*)d_in[18];
  const float* bnB    = (const float*)d_in[19];

  // ---- workspace carve (all offsets deterministic) ----
  char* wsp = (char*)d_ws;
  auto carve = [&](size_t bytes) -> void* {
    void* p = (void*)wsp;
    wsp += (bytes + 255) & ~(size_t)255;
    return p;
  };
  int*      idxall  = (int*)carve((size_t)4 * N_TOK * sizeof(int));
  float*    xT      = (float*)carve((size_t)N_TOK * CDIM * 4);
  _Float16* x16     = (_Float16*)carve((size_t)N_TOK * CDIM * 2);
  float*    qkv     = (float*)carve((size_t)N_TOK * QKVD * 4);   // also rqkv
  _Float16* qkv16   = (_Float16*)carve((size_t)N_TOK * QKVD * 2);// also rqkv16
  float*    attn    = (float*)carve((size_t)N_TOK * CDIM * 4);   // also rattn
  _Float16* tmp16   = (_Float16*)carve((size_t)N_TOK * FFND * 2);// o16/t16/hid16/r16
  float*    proj    = (float*)carve((size_t)N_TOK * CDIM * 4);   // also rproj
  float*    tbuf    = (float*)carve((size_t)N_TOK * CDIM * 4);
  float*    rbuf    = (float*)carve((size_t)512 * CDIM * 4);
  float*    r2buf   = (float*)carve((size_t)512 * CDIM * 4);
  float*    hid     = (float*)carve((size_t)N_TOK * FFND * 4);
  _Float16* fused16 = (_Float16*)carve((size_t)N_TOK * 4 * CDIM * 2);
  float*    ybuf    = (float*)carve((size_t)N_TOK * CDIM * 4);
  float*    stats   = (float*)carve((size_t)2 * CDIM * 4);
  _Float16* iWqkv16 = (_Float16*)carve((size_t)4 * QKVD * CDIM * 2);
  _Float16* iWout16 = (_Float16*)carve((size_t)4 * CDIM * CDIM * 2);
  _Float16* eWqkv16 = (_Float16*)carve((size_t)4 * QKVD * CDIM * 2);
  _Float16* eWout16 = (_Float16*)carve((size_t)4 * CDIM * CDIM * 2);
  _Float16* w1_16   = (_Float16*)carve((size_t)4 * FFND * CDIM * 2);
  _Float16* w2_16   = (_Float16*)carve((size_t)4 * CDIM * FFND * 2);
  _Float16* fuseW16 = (_Float16*)carve((size_t)CDIM * 4 * CDIM * 2);

  auto cvt = [&](const float* s, _Float16* d, int n) {
    k_f16<<<dim3((n + 255) / 256), dim3(256), 0, stream>>>(s, d, n);
  };
  auto gemm = [&](const _Float16* A, const _Float16* Bm, const float* bias,
                  const float* res, float* Cc, int M, int Nn, int K, int act) {
    k_gemm<<<dim3((M + 63) / 64, Nn / 64), dim3(128), 0, stream>>>(
        A, Bm, bias, res, Cc, M, Nn, K, act);
  };

  // ---- one-time prep ----
  cvt(iW_qkv, iWqkv16, 4 * QKVD * CDIM);
  cvt(iW_out, iWout16, 4 * CDIM * CDIM);
  cvt(eW_qkv, eWqkv16, 4 * QKVD * CDIM);
  cvt(eW_out, eWout16, 4 * CDIM * CDIM);
  cvt(fW1, w1_16, 4 * FFND * CDIM);
  cvt(fW2, w2_16, 4 * CDIM * FFND);
  cvt(fuseW, fuseW16, CDIM * 4 * CDIM);
  k_build_idx<<<dim3(64), dim3(256), 0, stream>>>(idxall);
  k_transpose<<<dim3(N_TOK), dim3(256), 0, stream>>>(X, xT);
  cvt(xT, x16, N_TOK * CDIM);

  const float scale = 0.17677669529663688f;   // 1/sqrt(32)

  for (int i = 0; i < 4; ++i) {
    int s = 1 << i;
    int bsz = 16 / s, nb = s * s * s, bs = bsz * bsz * bsz;
    const int* idx = idxall + i * N_TOK;

    // intra-block attention
    gemm(x16, iWqkv16 + (size_t)i * QKVD * CDIM, iB_qkv + i * QKVD, nullptr,
         qkv, N_TOK, QKVD, CDIM, 0);
    cvt(qkv, qkv16, N_TOK * QKVD);
    k_attn<<<dim3((bs + 15) / 16, NHEAD, nb), dim3(32), 0, stream>>>(
        qkv16, idx, attn, bs, scale);
    cvt(attn, tmp16, N_TOK * CDIM);
    gemm(tmp16, iWout16 + (size_t)i * CDIM * CDIM, iB_out + i * CDIM, nullptr,
         proj, N_TOK, CDIM, CDIM, 0);
    k_add_ln<<<dim3(N_TOK), dim3(256), 0, stream>>>(
        proj, xT, iLN_g + i * CDIM, iLN_b + i * CDIM, tbuf);

    // inter-block attention on block means
    k_block_mean<<<dim3(nb), dim3(256), 0, stream>>>(tbuf, idx, rbuf, bs);
    cvt(rbuf, tmp16, nb * CDIM);
    gemm(tmp16, eWqkv16 + (size_t)i * QKVD * CDIM, eB_qkv + i * QKVD, nullptr,
         qkv, nb, QKVD, CDIM, 0);
    cvt(qkv, qkv16, nb * QKVD);
    k_attn<<<dim3((nb + 15) / 16, NHEAD, 1), dim3(32), 0, stream>>>(
        qkv16, nullptr, attn, nb, scale);
    cvt(attn, tmp16, nb * CDIM);
    gemm(tmp16, eWout16 + (size_t)i * CDIM * CDIM, eB_out + i * CDIM, nullptr,
         proj, nb, CDIM, CDIM, 0);
    k_add_ln<<<dim3(nb), dim3(256), 0, stream>>>(
        proj, rbuf, eLN_g + i * CDIM, eLN_b + i * CDIM, r2buf);
    k_add_bcast<<<dim3(N_TOK), dim3(256), 0, stream>>>(tbuf, r2buf, idx, bs);

    // FFN with exact-GELU, residual
    cvt(tbuf, tmp16, N_TOK * CDIM);
    gemm(tmp16, w1_16 + (size_t)i * FFND * CDIM, fB1 + i * FFND, nullptr,
         hid, N_TOK, FFND, CDIM, 1);
    cvt(hid, tmp16, N_TOK * FFND);
    gemm(tmp16, w2_16 + (size_t)i * CDIM * FFND, fB2 + i * CDIM, tbuf,
         tbuf, N_TOK, CDIM, FFND, 0);

    // split residual + pack into fused channels [n, i*C + c]
    k_to_fused<<<dim3(N_TOK), dim3(256), 0, stream>>>(tbuf, xT, fused16, i * CDIM);
  }

  // 1x1x1 fuse conv: y[n, c] = fused[n, :] . fuseW[c, :]
  gemm(fused16, fuseW16, nullptr, nullptr, ybuf, N_TOK, CDIM, 4 * CDIM, 0);
  // BatchNorm (training-mode batch stats) + ReLU + transpose to [C, Z, H, W]
  k_bn_stats<<<dim3(CDIM), dim3(256), 0, stream>>>(ybuf, stats);
  k_bn_out<<<dim3(CDIM, N_TOK / 256), dim3(256), 0, stream>>>(
      ybuf, stats, bnG, bnB, (float*)d_out);
}
